// MambaActorCritic_87376814670398
// MI455X (gfx1250) — matrix operand order repaired and checked by reading, not compiled, not run on hardware
//
#include <hip/hip_runtime.h>
#include <hip/hip_bf16.h>
#include <math.h>

#define B_SZ     2048
#define OBS_N    512
#define D_MODEL  1024
#define D_STATE  16
#define D_CONV   4
#define D_INNER  2048
#define DT_RANK  64
#define ACT_N    18
#define XDB_COLS (DT_RANK + 2 * D_STATE)   // 96

typedef __attribute__((ext_vector_type(16))) __bf16 v16bf;
typedef __attribute__((ext_vector_type(8)))  float  v8f;

__device__ __forceinline__ float sigmoid_(float x) { return 1.0f / (1.0f + __expf(-x)); }
__device__ __forceinline__ float silu_(float x)    { return x * sigmoid_(x); }
__device__ __forceinline__ float softplus_(float x){ return (x > 20.0f) ? x : log1pf(__expf(x)); }

// ---------------------------------------------------------------------------
// B-matrix fragment (32x16, bf16), ISA 7.12.2 layout:
//   VGPR j, lanes 0-15  : K = 2j, 2j+1 (j<4), 8+2j, 9+2j (j>=4)   (column = lane)
//   VGPR j, lanes 16-31 : K = 8+2j, 9+2j (j<4), 16+2j, 17+2j (j>=4)
// B[k][n] = W[col0+n][k0+k]   (W stored [N,K] row-major -> W^T is the B operand)
// Per lane this is two runs of 8 f32: K = 8*half + {0..7} and 16 + 8*half + {0..7} (matches A).
// ---------------------------------------------------------------------------
__device__ __forceinline__ v16bf load_b_frag(const float* __restrict__ W, int ldw,
                                             int col0, int k0, int n, int half) {
  v16bf b;
  const float* wrow = W + (size_t)(col0 + n) * ldw + k0 + half * 8;
#pragma unroll
  for (int j = 0; j < 8; ++j) {
    b[2 * j + 0] = (__bf16)wrow[2 * j + (j >= 4 ? 8 : 0) + 0];
    b[2 * j + 1] = (__bf16)wrow[2 * j + (j >= 4 ? 8 : 0) + 1];
  }
  return b;
}

// D/C layout: lane -> N = lane&15; VGPR r -> M = r + 8*(lane>>4)
// For a fixed r, lanes 0..15 cover 16 consecutive N addresses -> coalesced.
__device__ __forceinline__ void store_tile(float* __restrict__ Y, int ldy, int row0, int col0,
                                           const float* __restrict__ bias, int act,
                                           v8f c, int n, int half) {
  float bv = bias ? bias[col0 + n] : 0.0f;
#pragma unroll
  for (int r = 0; r < 8; ++r) {
    float v = c[r] + bv;
    if (act == 1) v = softplus_(v);
    Y[(size_t)(row0 + r + half * 8) * ldy + (col0 + n)] = v;
  }
}

// ---------------------------------------------------------------------------
// Y[M,N] = act( A[M,K](lda) @ W[N,K](ldw)^T + bias[N] )
// One wave per 16(M) x 64(N) strip: A fragment loaded once per k-step,
// reused across 4 WMMAs (f32 accumulate, bf16 multiply).
// FULL_TILES=true: N is a multiple of 64 -> branch-free k-loop body
// (16x global_load_b128 + 32x v_cvt_pk_bf16_f32 + 4x v_wmma, no SALU guards).
// act: 0 = identity, 1 = softplus
// ---------------------------------------------------------------------------
template <bool FULL_TILES>
__global__ void __launch_bounds__(32)
wmma_gemm_kernel(const float* __restrict__ A, int lda,
                 const float* __restrict__ W, int ldw,
                 const float* __restrict__ bias,
                 float* __restrict__ Y, int ldy,
                 int M, int N, int K, int act) {
  const int lane    = threadIdx.x & 31;
  const int mrow    = lane & 15;
  const int half    = lane >> 4;
  const int row0    = blockIdx.y * 16;
  const int colbase = blockIdx.x * 64;
  if (row0 >= M) return;

  const bool v0 = FULL_TILES || (colbase +  0) < N;
  const bool v1 = FULL_TILES || (colbase + 16) < N;
  const bool v2 = FULL_TILES || (colbase + 32) < N;
  const bool v3 = FULL_TILES || (colbase + 48) < N;

  v8f c0 = {}, c1 = {}, c2 = {}, c3 = {};

  // A fragment (16x32 bf16): lanes 0-15: K in {0..7, 16..23}; lanes 16-31: +8
  const float* arow = A + (size_t)(row0 + mrow) * lda + (half ? 8 : 0);

  for (int k0 = 0; k0 < K; k0 += 32) {
    const float* ap = arow + k0;
    if (k0 + 32 < K) __builtin_prefetch(ap + 32, 0, 1);  // global_prefetch_b8

    v16bf a;
#pragma unroll
    for (int j = 0; j < 4; ++j) {
      a[2 * j + 0]     = (__bf16)ap[2 * j + 0];
      a[2 * j + 1]     = (__bf16)ap[2 * j + 1];
      a[8 + 2 * j + 0] = (__bf16)ap[16 + 2 * j + 0];
      a[8 + 2 * j + 1] = (__bf16)ap[16 + 2 * j + 1];
    }

    if (v0) {
      v16bf b = load_b_frag(W, ldw, colbase + 0, k0, mrow, half);
      c0 = __builtin_amdgcn_wmma_f32_16x16x32_bf16(false, a, false, b, (short)0, c0, false, false);
    }
    if (v1) {
      v16bf b = load_b_frag(W, ldw, colbase + 16, k0, mrow, half);
      c1 = __builtin_amdgcn_wmma_f32_16x16x32_bf16(false, a, false, b, (short)0, c1, false, false);
    }
    if (v2) {
      v16bf b = load_b_frag(W, ldw, colbase + 32, k0, mrow, half);
      c2 = __builtin_amdgcn_wmma_f32_16x16x32_bf16(false, a, false, b, (short)0, c2, false, false);
    }
    if (v3) {
      v16bf b = load_b_frag(W, ldw, colbase + 48, k0, mrow, half);
      c3 = __builtin_amdgcn_wmma_f32_16x16x32_bf16(false, a, false, b, (short)0, c3, false, false);
    }
  }

  if (v0) store_tile(Y, ldy, row0, colbase +  0, bias, act, c0, mrow, half);
  if (v1) store_tile(Y, ldy, row0, colbase + 16, bias, act, c1, mrow, half);
  if (v2) store_tile(Y, ldy, row0, colbase + 32, bias, act, c2, mrow, half);
  if (v3) store_tile(Y, ldy, row0, colbase + 48, bias, act, c3, mrow, half);
}

// ---------------------------------------------------------------------------
// Conv-state shift + depthwise conv + SiLU. One thread per (b,d).
// Reads conv_state (64 MB) once, writes conv_state_new (64 MB) once.
// ---------------------------------------------------------------------------
__global__ void conv_silu_kernel(const float* __restrict__ xz,
                                 const float* __restrict__ conv_state,
                                 const float* __restrict__ conv_w,
                                 const float* __restrict__ conv_b,
                                 float* __restrict__ conv_state_new,
                                 float* __restrict__ x_conv, int total) {
  int t = blockIdx.x * blockDim.x + threadIdx.x;
  if (t >= total) return;
  int b = t / D_INNER;
  int d = t - b * D_INNER;

  float xin = xz[(size_t)b * (2 * D_INNER) + d];
  const float* cs = conv_state + (size_t)t * D_CONV;
  float s0 = cs[1], s1 = cs[2], s2 = cs[3], s3 = xin;

  float* cn = conv_state_new + (size_t)t * D_CONV;
  cn[0] = s0; cn[1] = s1; cn[2] = s2; cn[3] = s3;

  const float* w = conv_w + (size_t)d * D_CONV;
  float acc = s0 * w[0] + s1 * w[1] + s2 * w[2] + s3 * w[3] + conv_b[d];
  x_conv[t] = silu_(acc);
}

// ---------------------------------------------------------------------------
// Selective-scan state update, fused with y = (C.state + D*x) * silu(z).
// One thread per (b,d); the 16-wide D_STATE loop streams ssm_state once
// (256 MB read + 256 MB write = the memory roofline of the whole net).
// ---------------------------------------------------------------------------
__global__ void ssm_update_kernel(const float* __restrict__ x_conv,
                                  const float* __restrict__ dt_full,
                                  const float* __restrict__ x_db,
                                  const float* __restrict__ xz,
                                  const float* __restrict__ A_log,
                                  const float* __restrict__ Dvec,
                                  const float* __restrict__ ssm_state,
                                  float* __restrict__ ssm_state_new,
                                  float* __restrict__ yout, int total) {
  int t = blockIdx.x * blockDim.x + threadIdx.x;
  if (t >= total) return;
  int b = t / D_INNER;
  int d = t - b * D_INNER;

  float dt = dt_full[t];
  float x  = x_conv[t];
  float z  = xz[(size_t)b * (2 * D_INNER) + D_INNER + d];

  const float* Bm = x_db + (size_t)b * XDB_COLS + DT_RANK;            // 16 floats
  const float* Cm = Bm + D_STATE;                                     // 16 floats
  const float* Al = A_log + (size_t)d * D_STATE;
  const float* st = ssm_state + (size_t)t * D_STATE;
  float* stn = ssm_state_new + (size_t)t * D_STATE;

  float xdt = x * dt;
  float y = 0.0f;
#pragma unroll
  for (int n = 0; n < D_STATE; ++n) {
    float dA = __expf(-__expf(Al[n]) * dt);   // A = -exp(A_log)
    float s  = st[n] * dA + xdt * Bm[n];
    stn[n] = s;
    y += s * Cm[n];
  }
  y += Dvec[d] * x;
  yout[t] = y * silu_(z);
}

// ---------------------------------------------------------------------------
// Policy/value heads: 19 length-1024 dot products per batch row.
// out-row cached in registers, shared-memory tree reduction.
// ---------------------------------------------------------------------------
__global__ void heads_kernel(const float* __restrict__ out,
                             const float* __restrict__ pol_w, const float* __restrict__ pol_b,
                             const float* __restrict__ val_w, const float* __restrict__ val_b,
                             float* __restrict__ logits, float* __restrict__ value) {
  int b = blockIdx.x, tid = threadIdx.x;     // 256 threads
  __shared__ float red[256];
  const float* orow = out + (size_t)b * D_MODEL;
  float o[4];
#pragma unroll
  for (int i = 0; i < 4; ++i) o[i] = orow[tid + i * 256];

  for (int j = 0; j < ACT_N + 1; ++j) {
    const float* w = (j < ACT_N) ? (pol_w + (size_t)j * D_MODEL) : val_w;
    float acc = 0.0f;
#pragma unroll
    for (int i = 0; i < 4; ++i) acc += o[i] * w[tid + i * 256];
    red[tid] = acc;
    __syncthreads();
    for (int s = 128; s > 0; s >>= 1) {
      if (tid < s) red[tid] += red[tid + s];
      __syncthreads();
    }
    if (tid == 0) {
      if (j < ACT_N) logits[(size_t)b * ACT_N + j] = red[0] + pol_b[j];
      else           value[b] = red[0] + val_b[0];
    }
    __syncthreads();
  }
}

// ---------------------------------------------------------------------------
extern "C" void kernel_launch(void* const* d_in, const int* in_sizes, int n_in,
                              void* d_out, int out_size, void* d_ws, size_t ws_size,
                              hipStream_t stream) {
  (void)in_sizes; (void)n_in; (void)out_size; (void)ws_size;

  const float* obs        = (const float*)d_in[0];
  const float* conv_state = (const float*)d_in[1];
  const float* ssm_state  = (const float*)d_in[2];
  const float* in_proj_w  = (const float*)d_in[3];
  const float* in_proj_b  = (const float*)d_in[4];
  const float* mamba_in_w = (const float*)d_in[5];
  const float* conv_w     = (const float*)d_in[6];
  const float* conv_b     = (const float*)d_in[7];
  const float* x_proj_w   = (const float*)d_in[8];
  const float* dt_w       = (const float*)d_in[9];
  const float* dt_b       = (const float*)d_in[10];
  const float* A_log      = (const float*)d_in[11];
  const float* Dvec       = (const float*)d_in[12];
  const float* out_proj_w = (const float*)d_in[13];
  const float* pol_w      = (const float*)d_in[14];
  const float* pol_b      = (const float*)d_in[15];
  const float* val_w      = (const float*)d_in[16];
  const float* val_b      = (const float*)d_in[17];

  // Output layout: logits | value | conv_state_new | ssm_state_new
  float* logits         = (float*)d_out;
  float* value          = logits + (size_t)B_SZ * ACT_N;
  float* conv_state_new = value + B_SZ;
  float* ssm_state_new  = conv_state_new + (size_t)B_SZ * D_INNER * D_CONV;

  // Workspace layout (~93 MB)
  float* ws      = (float*)d_ws;
  float* h       = ws;                                  // 2048*1024
  float* xz      = h       + (size_t)B_SZ * D_MODEL;    // 2048*4096
  float* x_conv  = xz      + (size_t)B_SZ * 2 * D_INNER;// 2048*2048
  float* x_db    = x_conv  + (size_t)B_SZ * D_INNER;    // 2048*96
  float* dt_full = x_db    + (size_t)B_SZ * XDB_COLS;   // 2048*2048
  float* ybuf    = dt_full + (size_t)B_SZ * D_INNER;    // 2048*2048
  float* outbuf  = h;  // h is dead after GEMM2; reuse its 8 MB for `out`

  const dim3 wave(32);
  const int total = B_SZ * D_INNER;

  // 1. h = obs @ in_proj_w^T + in_proj_b            [2048,1024]  (N%64==0 -> FULL)
  wmma_gemm_kernel<true><<<dim3(D_MODEL / 64, B_SZ / 16), wave, 0, stream>>>(
      obs, OBS_N, in_proj_w, OBS_N, in_proj_b, h, D_MODEL, B_SZ, D_MODEL, OBS_N, 0);

  // 2. xz = h @ mamba_in_w^T                        [2048,4096]  FULL
  wmma_gemm_kernel<true><<<dim3((2 * D_INNER) / 64, B_SZ / 16), wave, 0, stream>>>(
      h, D_MODEL, mamba_in_w, D_MODEL, nullptr, xz, 2 * D_INNER, B_SZ, 2 * D_INNER, D_MODEL, 0);

  // 3. conv shift + depthwise conv + SiLU           (writes conv_state_new)
  conv_silu_kernel<<<(total + 255) / 256, 256, 0, stream>>>(
      xz, conv_state, conv_w, conv_b, conv_state_new, x_conv, total);

  // 4. x_db = x_conv @ x_proj_w^T                   [2048,96]    N=96 -> guarded
  wmma_gemm_kernel<false><<<dim3((XDB_COLS + 63) / 64, B_SZ / 16), wave, 0, stream>>>(
      x_conv, D_INNER, x_proj_w, D_INNER, nullptr, x_db, XDB_COLS, B_SZ, XDB_COLS, D_INNER, 0);

  // 5. dt_full = softplus(x_db[:, :64] @ dt_w^T + dt_b)   [2048,2048]  FULL
  wmma_gemm_kernel<true><<<dim3(D_INNER / 64, B_SZ / 16), wave, 0, stream>>>(
      x_db, XDB_COLS, dt_w, DT_RANK, dt_b, dt_full, D_INNER, B_SZ, D_INNER, DT_RANK, 1);

  // 6. SSM state update + y                         (writes ssm_state_new)
  ssm_update_kernel<<<(total + 255) / 256, 256, 0, stream>>>(
      x_conv, dt_full, x_db, xz, A_log, Dvec, ssm_state, ssm_state_new, ybuf, total);

  // 7. out = y @ out_proj_w^T                       [2048,1024]  FULL
  wmma_gemm_kernel<true><<<dim3(D_MODEL / 64, B_SZ / 16), wave, 0, stream>>>(
      ybuf, D_INNER, out_proj_w, D_INNER, nullptr, outbuf, D_MODEL, B_SZ, D_MODEL, D_INNER, 0);

  // 8. heads: logits + value
  heads_kernel<<<B_SZ, 256, 0, stream>>>(outbuf, pol_w, pol_b, val_w, val_b, logits, value);
}
